// VectorQuantizer_26989574488266
// MI455X (gfx1250) — matrix operand order repaired and checked by reading, not compiled, not run on hardware
//
#include <hip/hip_runtime.h>

#define EMBED_DIM 64
#define NUM_CODES 128
#define N_ROWS (64 * 64 * 64)          // 262144 rows of x
#define ROWS_PER_TILE 16
#define N_TILES (N_ROWS / ROWS_PER_TILE) // 16384
#define WAVES_PER_BLOCK 8
#define OUT_ELEMS (N_ROWS * EMBED_DIM)   // 16777216

typedef __attribute__((ext_vector_type(16))) __bf16 v16bf;
typedef __attribute__((ext_vector_type(8)))  float  v8f;

// ---------------------------------------------------------------------------
// Prep: build code-major codebook copies (f32 for gather, bf16 for WMMA B),
// per-code squared norms, and zero the loss accumulator.
// w is [64][128] row-major; wT* are [128][64] code-major.
// ---------------------------------------------------------------------------
__global__ __launch_bounds__(NUM_CODES)
void vq_prep_kernel(const float* __restrict__ w,
                    float* __restrict__ wTf32,
                    __bf16* __restrict__ wTbf,
                    float* __restrict__ wnorm,
                    float* __restrict__ loss) {
    int k = threadIdx.x;                 // code index 0..127
    float nrm = 0.0f;
#pragma unroll 8
    for (int d = 0; d < EMBED_DIM; ++d) {
        float v = w[d * NUM_CODES + k];
        wTf32[k * EMBED_DIM + d] = v;
        wTbf[k * EMBED_DIM + d]  = (__bf16)v;
        nrm += v * v;
    }
    wnorm[k] = nrm;
    if (k == 0) *loss = 0.0f;
}

// ---------------------------------------------------------------------------
// Main: one wave32 handles a 16-row tile of x against all 128 codes.
//   score(row, code) = ||w_code||^2 - 2 * <x_row, w_code>
// Dot products via v_wmma_f32_16x16x32_bf16 (K=64 in 2 steps, f32 acc).
// ---------------------------------------------------------------------------
__global__ __launch_bounds__(32 * WAVES_PER_BLOCK)
void vq_main_kernel(const float* __restrict__ x,
                    const float* __restrict__ wTf32,
                    const __bf16* __restrict__ wTbf,
                    const float* __restrict__ wnorm,
                    float* __restrict__ out,
                    float* __restrict__ loss) {
    const int lane = threadIdx.x & 31;
    const int wid  = (blockIdx.x * blockDim.x + threadIdx.x) >> 5; // tile id
    const int hi   = lane >> 4;      // half-wave select
    const int col  = lane & 15;      // 0..15

    const long row0 = (long)wid * ROWS_PER_TILE;

    // ---------------- A-matrix tile: 16 rows x K=64 of x, bf16 --------------
    // 16-bit A 16x32 layout used here (contiguous k, not ISA table order): lane (both halves) holds row M = lane%16.
//   elements e=0..15 -> K = kb + hi*16 + e
    const float* xr = x + (row0 + col) * EMBED_DIM;
    v16bf a0, a1;
#pragma unroll
    for (int kc = 0; kc < 2; ++kc) {
        const int kb = kc * 32;
        float4 f0 = *(const float4*)(xr + kb + hi * 16 + 0);
        float4 f1 = *(const float4*)(xr + kb + hi * 16 + 4);
        float4 f2 = *(const float4*)(xr + kb + hi * 16 + 8);
        float4 f3 = *(const float4*)(xr + kb + hi * 16 + 12);
        v16bf av;
        av[0]  = (__bf16)f0.x; av[1]  = (__bf16)f0.y; av[2]  = (__bf16)f0.z; av[3]  = (__bf16)f0.w;
        av[4]  = (__bf16)f1.x; av[5]  = (__bf16)f1.y; av[6]  = (__bf16)f1.z; av[7]  = (__bf16)f1.w;
        av[8]  = (__bf16)f2.x; av[9]  = (__bf16)f2.y; av[10] = (__bf16)f2.z; av[11] = (__bf16)f2.w;
        av[12] = (__bf16)f3.x; av[13] = (__bf16)f3.y; av[14] = (__bf16)f3.z; av[15] = (__bf16)f3.w;
        if (kc == 0) a0 = av; else a1 = av;
    }

    // ---------------- scan 8 code tiles of 16 codes each --------------------
    // D-matrix 16x16 f32 layout: lane holds column N = col; VGPR j holds row
    // M = hi*8 + j. So per lane: 8 rows x 1 code.
    float bestv[8];
    int   besti[8];
#pragma unroll
    for (int j = 0; j < 8; ++j) { bestv[j] = 3.4e38f; besti[j] = 0; }

#pragma unroll
    for (int c = 0; c < 8; ++c) {
        const int code = c * 16 + col;
        // B 32x16 bf16 layout: lane col = column, lane half selects K block of
        // 16; code-major bf16 codebook makes this one contiguous 32B chunk.
        const v16bf b0 = *(const v16bf*)(wTbf + code * EMBED_DIM + 0  + hi * 16);
        const v16bf b1 = *(const v16bf*)(wTbf + code * EMBED_DIM + 32 + hi * 16);
        v8f acc = {};
        acc = __builtin_amdgcn_wmma_f32_16x16x32_bf16(false, a0, false, b0,
                                                      (short)0, acc, false, false);
        acc = __builtin_amdgcn_wmma_f32_16x16x32_bf16(false, a1, false, b1,
                                                      (short)0, acc, false, false);
        const float wn = wnorm[code];
#pragma unroll
        for (int j = 0; j < 8; ++j) {
            float s = wn - 2.0f * acc[j];
            if (s < bestv[j]) { bestv[j] = s; besti[j] = code; }
        }
    }

    // ---------------- argmin reduce across the 16 lanes of each half --------
#pragma unroll
    for (int off = 8; off >= 1; off >>= 1) {
#pragma unroll
        for (int j = 0; j < 8; ++j) {
            float ov = __shfl_xor(bestv[j], off, 32);
            int   oi = __shfl_xor(besti[j], off, 32);
            if (ov < bestv[j] || (ov == bestv[j] && oi < besti[j])) {
                bestv[j] = ov; besti[j] = oi;
            }
        }
    }
    // Now every lane of half 'hi' holds the winning code for rows hi*8+j.

    // ---------------- gather + store + loss ---------------------------------
    float lsum = 0.0f;
#pragma unroll
    for (int j = 0; j < 8; ++j) {
        const long row  = row0 + hi * 8 + j;
        const int  code = besti[j];
        float4 qv = *(const float4*)(wTf32 + code * EMBED_DIM + col * 4);
        float4 xv = *(const float4*)(x + row * EMBED_DIM + col * 4);
        *(float4*)(out + row * EMBED_DIM + col * 4) = qv;
        float d0 = qv.x - xv.x, d1 = qv.y - xv.y;
        float d2 = qv.z - xv.z, d3 = qv.w - xv.w;
        lsum += d0 * d0 + d1 * d1 + d2 * d2 + d3 * d3;
    }
#pragma unroll
    for (int off = 16; off >= 1; off >>= 1)
        lsum += __shfl_xor(lsum, off, 32);
    if (lane == 0) {
        // loss = commitment + codebook = 2 * mean((q-x)^2)
        atomicAdd(loss, lsum * (2.0f / (float)OUT_ELEMS));
    }
}

// ---------------------------------------------------------------------------
extern "C" void kernel_launch(void* const* d_in, const int* in_sizes, int n_in,
                              void* d_out, int out_size, void* d_ws, size_t ws_size,
                              hipStream_t stream) {
    const float* x = (const float*)d_in[0];  // [64,64,64,64] f32
    const float* w = (const float*)d_in[1];  // [64,128] f32
    float* out  = (float*)d_out;             // OUT_ELEMS floats + 1 loss float
    float* loss = out + (out_size - 1);

    char* ws = (char*)d_ws;
    float*  wTf32 = (float*)ws;                          // 128*64*4 = 32768 B
    __bf16* wTbf  = (__bf16*)(ws + 32768);               // 128*64*2 = 16384 B
    float*  wnorm = (float*)(ws + 32768 + 16384);        // 128*4    =   512 B

    vq_prep_kernel<<<1, NUM_CODES, 0, stream>>>(w, wTf32, wTbf, wnorm, loss);

    const int blocks = N_TILES / WAVES_PER_BLOCK;        // 2048
    vq_main_kernel<<<blocks, 32 * WAVES_PER_BLOCK, 0, stream>>>(
        x, wTf32, wTbf, wnorm, out, loss);
}
